// LSTMModule_901943132624
// MI455X (gfx1250) — matrix operand order repaired and checked by reading, not compiled, not run on hardware
//
#include <hip/hip_runtime.h>
#include <math.h>

// ---------------- problem constants ----------------
constexpr int B  = 256;
constexpr int T  = 512;
constexpr int F  = 64;
constexpr int H1 = 512;
constexpr int H2 = 256;

// ---------------- WMMA types ----------------
typedef __bf16 bf16_t;
typedef __attribute__((ext_vector_type(16))) __bf16        v16bf;
typedef __attribute__((ext_vector_type(8)))  float         v8f;
typedef __attribute__((ext_vector_type(4)))  unsigned int  u32x4;

union Frag { v16bf v; u32x4 q[2]; };

__device__ __forceinline__ float sigmoid_f(float x) {
    return 1.0f / (1.0f + __expf(-x));
}

// ---------------- prep: x -> bf16 ----------------
__global__ void cvt_x_bf16(const float* __restrict__ x, bf16_t* __restrict__ xb, int n) {
    int i = blockIdx.x * 256 + threadIdx.x;
    if (i < n) xb[i] = (bf16_t)x[i];
}

// ---------------- prep: pack weights into WMMA B-fragment layout ----------------
// B fragment (16x16x32 bf16): lane l holds N = nt*16 + (l&15); element j holds K = kt*32 + 8*(l>>4) + (j<8 ? j : 8+j).
// Element index tid == (((g*NT + nt)*KT + kt)*512 + lane*16 + j) -> destination is identity.
template<int H, int FIN>
__global__ void pack_w(const float* __restrict__ Wx, const float* __restrict__ Wh,
                       bf16_t* __restrict__ pw) {
    constexpr int K  = FIN + H;
    constexpr int KT = K / 32;
    constexpr int NT = H / 16;
    constexpr int TOT = 4 * H * K;
    int tid = blockIdx.x * 256 + threadIdx.x;
    if (tid >= TOT) return;
    int j    = tid & 15;
    int lane = (tid >> 4) & 31;
    int kt   = (tid >> 9) % KT;
    int rest = (tid >> 9) / KT;
    int nt   = rest % NT;
    int g    = rest / NT;
    int k = kt * 32 + 8 * (lane >> 4) + (j < 8 ? j : 8 + j); // K index in concat [input ; hidden]
    int n = nt * 16 + (lane & 15);         // column within gate
    int col = g * H + n;                   // column in (K, 4H) weight
    float v = (k < FIN) ? Wx[(size_t)k * (4 * H) + col]
                        : Wh[(size_t)(k - FIN) * (4 * H) + col];
    pw[tid] = (bf16_t)v;
}

// ---------------- prep: zero initial state ----------------
__global__ void zero_state(float* __restrict__ c1, float* __restrict__ c2,
                           bf16_t* __restrict__ h1, bf16_t* __restrict__ h2) {
    int i = blockIdx.x * 256 + threadIdx.x;
    if (i < B * H1) { c1[i] = 0.0f; h1[i] = (bf16_t)0.0f; }
    if (i < B * H2) { c2[i] = 0.0f; h2[i] = (bf16_t)0.0f; }
}

// ---------------- fused LSTM step: gate GEMM (WMMA) + cell update ----------------
// One wave per block. Each wave owns a 32-row x 16-col tile of h (2 m-tiles),
// computes i/f/g/o gate tiles for it (4 gates x 2 m-tiles = 8 fp32 accumulators;
// each B fragment feeds 2 WMMAs), with an explicit double-buffered K pipeline:
// k-tile kt+1's loads are issued before k-tile kt's WMMA batch executes.
// Afterwards applies the nonlinearities and writes c (f32) and h_new (bf16).
// A = concat([in0 (B x FIN, row stride in0_stride), h_prev (B x H)]) in bf16.
template<int H, int FIN>
__global__ __launch_bounds__(32)
void lstm_step(const bf16_t* __restrict__ in0, int in0_stride,
               const bf16_t* __restrict__ hprev,
               const bf16_t* __restrict__ pw,
               const float*  __restrict__ bias,
               float*        __restrict__ c,
               bf16_t*       __restrict__ hnew) {
    constexpr int K   = FIN + H;
    constexpr int KT  = K / 32;
    constexpr int KT0 = FIN / 32;
    constexpr int NT  = H / 16;
    constexpr int MS  = B / 32;               // m-supertiles (32 rows each) == 8
    static_assert(K % 32 == 0 && H % 16 == 0 && B % 32 == 0, "tile mismatch");

    const int lane  = threadIdx.x;
    const int msup  = blockIdx.x & (MS - 1);  // 32-row supertile
    const int ntile = blockIdx.x / MS;        // h-column tile

    // A-fragment addressing (16-bit A 16x32): lane = {m, k-half}; per lane two b128 loads
    const int am   = lane & 15;
    const int akh  = lane >> 4;               // 0: K 0..7 / 16..23, 1: K 8..15 / 24..31
    const int row0 = msup * 32 + am;          // m-tile 0
    const int row1 = row0 + 16;               // m-tile 1

    // double-buffered fragments
    Frag afr[2][2];                           // [buf][m-tile]
    Frag bfr[2][4];                           // [buf][gate]

    auto loadA = [&](int buf, int kt) {
        const bf16_t *ap0, *ap1;
        if (kt < KT0) {
            const int kc = kt * 32 + akh * 8;
            ap0 = in0 + (size_t)row0 * in0_stride + kc;
            ap1 = in0 + (size_t)row1 * in0_stride + kc;
        } else {
            const int kc = (kt - KT0) * 32 + akh * 8;
            ap0 = hprev + (size_t)row0 * H + kc;
            ap1 = hprev + (size_t)row1 * H + kc;
        }
        afr[buf][0].q[0] = *(const u32x4*)(ap0);
        afr[buf][0].q[1] = *(const u32x4*)(ap0 + 16);
        afr[buf][1].q[0] = *(const u32x4*)(ap1);
        afr[buf][1].q[1] = *(const u32x4*)(ap1 + 16);
    };
    auto loadB = [&](int buf, int kt) {
        const bf16_t* bp = pw + ((size_t)ntile * KT + kt) * 512 + lane * 16;
        #pragma unroll
        for (int g = 0; g < 4; ++g) {
            const bf16_t* bgp = bp + (size_t)g * NT * KT * 512;
            bfr[buf][g].q[0] = *(const u32x4*)(bgp);
            bfr[buf][g].q[1] = *(const u32x4*)(bgp + 8);
        }
    };

    v8f acc[4][2];
    #pragma unroll
    for (int g = 0; g < 4; ++g)
        #pragma unroll
        for (int mt = 0; mt < 2; ++mt)
            #pragma unroll
            for (int r = 0; r < 8; ++r) acc[g][mt][r] = 0.0f;

    // pipeline prologue
    loadA(0, 0);
    loadB(0, 0);

    #pragma unroll
    for (int kt = 0; kt < KT; ++kt) {
        const int cur = kt & 1, nxt = cur ^ 1;
        if (kt + 1 < KT) {                    // prefetch next k-tile before computing
            loadA(nxt, kt + 1);
            loadB(nxt, kt + 1);
        }
        #pragma unroll
        for (int g = 0; g < 4; ++g) {
            acc[g][0] = __builtin_amdgcn_wmma_f32_16x16x32_bf16(
                false, afr[cur][0].v, false, bfr[cur][g].v, (short)0, acc[g][0], false, false);
            acc[g][1] = __builtin_amdgcn_wmma_f32_16x16x32_bf16(
                false, afr[cur][1].v, false, bfr[cur][g].v, (short)0, acc[g][1], false, false);
        }
    }

    // ---- cell update. C/D layout: lanes 0-15 -> M = r, lanes 16-31 -> M = r+8, N = lane&15
    const int ncol = lane & 15;
    const int col  = ntile * 16 + ncol;
    const float bi = bias[0 * H + col];
    const float bf_ = bias[1 * H + col];
    const float bg = bias[2 * H + col];
    const float bo = bias[3 * H + col];
    #pragma unroll
    for (int mt = 0; mt < 2; ++mt) {
        const int rowbase = msup * 32 + mt * 16 + (akh ? 8 : 0);
        #pragma unroll
        for (int r = 0; r < 8; ++r) {
            const int row = rowbase + r;
            const size_t off = (size_t)row * H + col;
            const float iv = sigmoid_f(acc[0][mt][r] + bi);
            const float fv = sigmoid_f(acc[1][mt][r] + bf_);
            const float gv = tanhf(acc[2][mt][r] + bg);
            const float ov = sigmoid_f(acc[3][mt][r] + bo);
            const float cn = fv * c[off] + iv * gv;
            c[off] = cn;
            hnew[off] = (bf16_t)(ov * tanhf(cn));
        }
    }
}

// ---------------- dense head: y = h2_last @ Wd + bd ----------------
__global__ void dense_head(const bf16_t* __restrict__ h2,
                           const float* __restrict__ Wd,
                           const float* __restrict__ bd,
                           float* __restrict__ out) {
    int m = threadIdx.x;                      // 256 threads, B rows
    float s = 0.0f;
    #pragma unroll 8
    for (int k = 0; k < H2; ++k) s += (float)h2[(size_t)m * H2 + k] * Wd[k];
    out[m] = s + bd[0];
}

// ---------------- launch ----------------
extern "C" void kernel_launch(void* const* d_in, const int* in_sizes, int n_in,
                              void* d_out, int out_size, void* d_ws, size_t ws_size,
                              hipStream_t stream) {
    const float* x   = (const float*)d_in[0];
    const float* Wx1 = (const float*)d_in[1];
    const float* Wh1 = (const float*)d_in[2];
    const float* b1  = (const float*)d_in[3];
    const float* Wx2 = (const float*)d_in[4];
    const float* Wh2 = (const float*)d_in[5];
    const float* b2  = (const float*)d_in[6];
    const float* Wd  = (const float*)d_in[7];
    const float* bd  = (const float*)d_in[8];
    float* out = (float*)d_out;

    // workspace carve (256B aligned), total ~22 MB
    char* w = (char*)d_ws;
    auto carve = [&](size_t bytes) {
        char* p = w; w += (bytes + 255) & ~(size_t)255; return p;
    };
    bf16_t* xb   = (bf16_t*)carve((size_t)B * T * F * 2);
    bf16_t* pw1  = (bf16_t*)carve((size_t)4 * H1 * (F  + H1) * 2);
    bf16_t* pw2  = (bf16_t*)carve((size_t)4 * H2 * (H1 + H2) * 2);
    bf16_t* h1b[2] = { (bf16_t*)carve((size_t)B * H1 * 2), (bf16_t*)carve((size_t)B * H1 * 2) };
    bf16_t* h2b[2] = { (bf16_t*)carve((size_t)B * H2 * 2), (bf16_t*)carve((size_t)B * H2 * 2) };
    float*  c1   = (float*)carve((size_t)B * H1 * 4);
    float*  c2   = (float*)carve((size_t)B * H2 * 4);

    // prep
    {
        int n = B * T * F;
        cvt_x_bf16<<<(n + 255) / 256, 256, 0, stream>>>(x, xb, n);
        int tot1 = 4 * H1 * (F + H1);
        pack_w<H1, F ><<<(tot1 + 255) / 256, 256, 0, stream>>>(Wx1, Wh1, pw1);
        int tot2 = 4 * H2 * (H1 + H2);
        pack_w<H2, H1><<<(tot2 + 255) / 256, 256, 0, stream>>>(Wx2, Wh2, pw2);
        zero_state<<<(B * H1 + 255) / 256, 256, 0, stream>>>(c1, c2, h1b[0], h2b[0]);
    }

    // recurrence: per-step kernels give grid-wide sync; h double-buffered
    for (int t = 0; t < T; ++t) {
        const int cur = t & 1, nxt = cur ^ 1;
        lstm_step<H1, F ><<<(B / 32) * (H1 / 16), 32, 0, stream>>>(
            xb + (size_t)t * F, T * F, h1b[cur], pw1, b1, c1, h1b[nxt]);
        lstm_step<H2, H1><<<(B / 32) * (H2 / 16), 32, 0, stream>>>(
            h1b[nxt], H1, h2b[cur], pw2, b2, c2, h2b[nxt]);
    }

    const int last = ((T - 1) & 1) ^ 1;       // buffer holding h2 at t = T-1
    dense_head<<<1, B, 0, stream>>>(h2b[last], Wd, bd, out);
    (void)in_sizes; (void)n_in; (void)out_size; (void)ws_size;
}
